// MLA_21363167330627
// MI455X (gfx1250) — matrix operand order repaired and checked by reading, not compiled, not run on hardware
//
#include <hip/hip_runtime.h>
#include <hip/hip_bf16.h>
#include <stdint.h>

// ---------------- problem constants ----------------
#define B_    2
#define S_    2048
#define DIM_  2048
#define NH_   16
#define QLR_  1536
#define KVLR_ 512
#define DN_   128
#define DR_   64
#define DQK_  192   // DN + DR
#define DV_   128
#define EPS_  1e-6f
#define MTOK  (B_*S_)   // 4096 tokens

typedef __attribute__((ext_vector_type(16))) __bf16 v16bf;
typedef __attribute__((ext_vector_type(8)))  float  v8f;

union Frag16 { v16bf v; uint32_t u[8]; };

#ifndef __has_builtin
#define __has_builtin(x) 0
#endif
#if __has_builtin(__builtin_amdgcn_global_load_async_to_lds_b128)
#define HAVE_ASYNC_LDS 1
#else
#define HAVE_ASYNC_LDS 0
#endif

// pointer types matching the async-load builtin's signature:
// param0: v4i in addrspace(1) (global), param1: v4i in addrspace(3) (LDS)
typedef int v4i_ __attribute__((vector_size(16)));
typedef __attribute__((address_space(1))) v4i_* gptr_v4i;
typedef __attribute__((address_space(3))) v4i_* lptr_v4i;

// 16B global -> LDS copy; async (ASYNCcnt-tracked) when available.
__device__ __forceinline__ void cp16_to_lds(void* lds_dst, const void* gsrc) {
#if HAVE_ASYNC_LDS
  __builtin_amdgcn_global_load_async_to_lds_b128(
      (gptr_v4i)(uintptr_t)gsrc,
      (lptr_v4i)(uint32_t)(uintptr_t)lds_dst,
      0, 0);
#else
  *(uint4*)lds_dst = *(const uint4*)gsrc;
#endif
}

#if HAVE_ASYNC_LDS
#if __has_builtin(__builtin_amdgcn_s_wait_asynccnt)
#define WAIT_ASYNC(n) __builtin_amdgcn_s_wait_asynccnt(n)
#else
#define WAIT_ASYNC(n) asm volatile("s_wait_asynccnt %0" ::"i"(n) : "memory")
#endif
#else
#define WAIT_ASYNC(n) ((void)0)
#endif

// round-to-nearest-even f32 -> bf16 (bit pattern)
__device__ __forceinline__ uint16_t f2bf(float f) {
  uint32_t u = __float_as_uint(f);
  u += 0x7FFFu + ((u >> 16) & 1u);
  return (uint16_t)(u >> 16);
}

// ---------------- elementwise f32 -> bf16 ----------------
__global__ void k_cvt_bf16(const float* __restrict__ in, uint16_t* __restrict__ out, int n) {
  int stride = gridDim.x * blockDim.x;
  for (int i = blockIdx.x * blockDim.x + threadIdx.x; i < n; i += stride)
    out[i] = f2bf(in[i]);
}

// ---------------- generic bf16 WMMA GEMM: C[M,N] = A[M,K] * W[N,K]^T + bias ----------------
// 256 threads = 8 waves (2 M x 4 N); block tile 128x64; K step 32.
// Double-buffered LDS tiles filled with async global->LDS copies.
__launch_bounds__(256)
__global__ void k_gemm_bf16(const uint16_t* __restrict__ A, const uint16_t* __restrict__ W,
                            const float* __restrict__ bias, float* __restrict__ C,
                            int M, int N, int K) {
  __shared__ alignas(16) uint16_t sA[2][128 * 32];
  __shared__ alignas(16) uint16_t sB[2][64 * 32];
  const int tid  = threadIdx.x;
  const int lane = tid & 31, w = tid >> 5;
  const int m_l  = lane & 15, half = lane >> 4;
  const int wmy  = w >> 2,   wnx  = w & 3;
  const int rowBase = blockIdx.y * 128;
  const int colBase = blockIdx.x * 64;

  // stage tile at k0 into buffer bb: A 512 x 16B chunks (2/thread), W 256 x 16B (1/thread)
  auto stage = [&](int k0, int bb) {
#pragma unroll
    for (int j = 0; j < 2; ++j) {
      int c = j * 256 + tid;
      int row = c >> 2, part = c & 3;
      cp16_to_lds(&sA[bb][c * 8], A + (size_t)(rowBase + row) * K + k0 + part * 8);
    }
    {
      int row = tid >> 2, part = tid & 3;
      cp16_to_lds(&sB[bb][tid * 8], W + (size_t)(colBase + row) * K + k0 + part * 8);
    }
  };

  __builtin_prefetch(A + (size_t)rowBase * K, 0, 1);
  __builtin_prefetch(W + (size_t)colBase * K, 0, 1);

  v8f acc[4] = {};
  const int nk = K >> 5;
  stage(0, 0);

  for (int kk = 0; kk < nk; ++kk) {
    const int cur = kk & 1;
    if (kk + 1 < nk) { stage((kk + 1) << 5, cur ^ 1); WAIT_ASYNC(3); }
    else             { WAIT_ASYNC(0); }
    __syncthreads();

    Frag16 bfr;
#pragma unroll
    for (int v = 0; v < 8; ++v)   // B 32x16 bf16: K = (v/4)*16 + half*8 + (v%4)*2, +1
      bfr.u[v] = *(const uint32_t*)(&sB[cur][(wnx * 16 + m_l) * 32 + ((v >> 2) * 16 + half * 8 + (v & 3) * 2)]);

#pragma unroll
    for (int i = 0; i < 4; ++i) {
      Frag16 afr;
      int row = wmy * 64 + i * 16 + m_l;
#pragma unroll
      for (int v = 0; v < 8; ++v) // A 16x32 bf16: K = (v/4)*16 + half*8 + (v%4)*2
        afr.u[v] = *(const uint32_t*)(&sA[cur][row * 32 + ((v >> 2) * 16 + half * 8 + (v & 3) * 2)]);
      acc[i] = __builtin_amdgcn_wmma_f32_16x16x32_bf16(false, afr.v, false, bfr.v,
                                                       (short)0, acc[i], false, false);
    }
    __syncthreads();  // all waves done with buffer `cur` before it is refilled
  }

#pragma unroll
  for (int i = 0; i < 4; ++i) {
    int grow = rowBase + wmy * 64 + i * 16 + half * 8; // D row = r + 8*half
    int gcol = colBase + wnx * 16 + m_l;               // D col = lane%16
    float bv = bias ? bias[gcol] : 0.f;
#pragma unroll
    for (int r = 0; r < 8; ++r)
      C[(size_t)(grow + r) * N + gcol] = acc[i][r] + bv;
  }
}

// ---------------- RMSNorm rows: f32 in (stride in_stride), bf16 out (len L) ----------------
__global__ void k_rmsnorm_bf16(const float* __restrict__ in, const float* __restrict__ wgt,
                               uint16_t* __restrict__ out, int L, int in_stride) {
  __shared__ float red[256];
  const int row = blockIdx.x;
  const float* x = in + (size_t)row * in_stride;
  float ss = 0.f;
  for (int c = threadIdx.x; c < L; c += 256) { float v = x[c]; ss += v * v; }
  red[threadIdx.x] = ss; __syncthreads();
  for (int s = 128; s > 0; s >>= 1) {
    if (threadIdx.x < s) red[threadIdx.x] += red[threadIdx.x + s];
    __syncthreads();
  }
  float scale = rsqrtf(red[0] / (float)L + EPS_);
  for (int c = threadIdx.x; c < L; c += 256)
    out[(size_t)row * L + c] = f2bf(x[c] * scale * wgt[c]);
}

// RoPE on a 64-vector at `base`, position `pos`
__device__ __forceinline__ float rope_val(const float* base, int j, float pos) {
  if (j < 32) {
    float f = pos * __powf(10000.f, -(float)(2 * j) / 64.f);
    return base[j] * __cosf(f) - base[j + 32] * __sinf(f);
  } else {
    int i = j - 32;
    float f = pos * __powf(10000.f, -(float)(2 * i) / 64.f);
    return base[j] * __cosf(f) + base[i] * __sinf(f);
  }
}

// ---------------- q (f32 [tok, NH*192]) -> qf (bf16 [tok, h, 192]) with RoPE on last 64 ----
__global__ void k_rope_assemble_q(const float* __restrict__ q, uint16_t* __restrict__ qf) {
  const int tok = blockIdx.x;
  const float pos = (float)(tok & (S_ - 1));
  const float* qrow = q + (size_t)tok * (NH_ * DQK_);
  uint16_t* orow = qf + (size_t)tok * (NH_ * DQK_);
  for (int idx = threadIdx.x; idx < NH_ * DQK_; idx += blockDim.x) {
    int h = idx / DQK_, d = idx - h * DQK_;
    float v = (d < DN_) ? qrow[idx] : rope_val(qrow + h * DQK_ + DN_, d - DN_, pos);
    orow[idx] = f2bf(v);
  }
}

// ---------------- k_pe: rope(kv[:,512:576]) -> bf16 [tok,64] ----------------
__global__ void k_rope_kpe(const float* __restrict__ kv, uint16_t* __restrict__ kpe) {
  int idx = blockIdx.x * blockDim.x + threadIdx.x;
  if (idx >= MTOK * DR_) return;
  int tok = idx / DR_, j = idx - tok * DR_;
  float pos = (float)(tok & (S_ - 1));
  kpe[idx] = f2bf(rope_val(kv + (size_t)tok * (KVLR_ + DR_) + KVLR_, j, pos));
}

// --------- kb (f32 [tok, NH*256]) + kpe -> kf (bf16 [tok,h,192]), vT (bf16 [b*NH][d][S]) ----
__global__ void k_assemble_kv(const float* __restrict__ kb, const uint16_t* __restrict__ kpe,
                              uint16_t* __restrict__ kf, uint16_t* __restrict__ vT) {
  int idx = blockIdx.x * blockDim.x + threadIdx.x;
  if (idx >= MTOK * NH_ * DN_) return;
  int d = idx & 127;
  int h = (idx >> 7) & 15;
  int tok = idx >> 11;
  int bh = (tok / S_) * NH_ + h;
  int s  = tok & (S_ - 1);
  const float* src = kb + (size_t)tok * (NH_ * (DN_ + DV_)) + h * (DN_ + DV_);
  kf[((size_t)tok * NH_ + h) * DQK_ + d] = f2bf(src[d]);
  vT[((size_t)bh * DV_ + d) * S_ + s]    = f2bf(src[DN_ + d]);   // transposed V
  if (d < DR_) kf[((size_t)tok * NH_ + h) * DQK_ + DN_ + d] = kpe[tok * DR_ + d];
}

// ---------------- flash attention: 1 block per (b, h, 64-query tile); 4 waves x 16 rows ----
// K tile (32x192) and V^T tile (128x32) staged to LDS (async, double-buffered), shared by
// all 4 waves. P re-layout D->A goes through per-wave LDS scratch.
__launch_bounds__(128)
__global__ void k_attention(const uint16_t* __restrict__ qf, const uint16_t* __restrict__ kf,
                            const uint16_t* __restrict__ vT, uint16_t* __restrict__ ao) {
  __shared__ alignas(16) uint16_t ldsK[2][32 * DQK_];   // [token][192]
  __shared__ alignas(16) uint16_t ldsV[2][DV_ * 32];    // [d][token]
  __shared__ alignas(4)  uint16_t ldsP[4][16][32];
  const int tid = threadIdx.x;
  const int lane = tid & 31, w = tid >> 5;
  const int m_l = lane & 15, half = lane >> 4;
  const int qt = blockIdx.x & 31;
  const int h  = (blockIdx.x >> 5) & 15;
  const int b  = blockIdx.x >> 9;
  const int bh = b * NH_ + h;
  const int q0 = qt * 64 + w * 16;
  const size_t tq = (size_t)(b * S_ + q0 + m_l);
  const float sc = 0.07216878364870323f;  // 1/sqrt(192)

  // stage K (768 x 16B, 6/thread) and V^T (512 x 16B, 4/thread) tiles for kv step t0
  auto stage = [&](int t0, int bb) {
#pragma unroll
    for (int j = 0; j < 6; ++j) {
      int c = j * 128 + tid;
      int row = c / 24, col8 = c % 24;
      cp16_to_lds(&ldsK[bb][c * 8],
                  kf + ((size_t)(b * S_ + t0 + row) * NH_ + h) * DQK_ + col8 * 8);
    }
#pragma unroll
    for (int j = 0; j < 4; ++j) {
      int c = j * 128 + tid;
      int d = c >> 2, part = c & 3;
      cp16_to_lds(&ldsV[bb][c * 8],
                  vT + ((size_t)bh * DV_ + d) * S_ + t0 + part * 8);
    }
  };

  Frag16 qfr[6];  // Q row-tile 16x192 as 6 A-fragments (K chunks of 32)
#pragma unroll
  for (int c = 0; c < 6; ++c)
#pragma unroll
    for (int v = 0; v < 8; ++v) {
      int kidx = c * 32 + (v >> 2) * 16 + half * 8 + (v & 3) * 2;
      qfr[c].u[v] = *(const uint32_t*)(qf + (tq * NH_ + h) * DQK_ + kidx);
    }

  v8f o[8] = {};
  float m_run[8], l_run[8];
#pragma unroll
  for (int r = 0; r < 8; ++r) { m_run[r] = -3.0e38f; l_run[r] = 0.f; }

  stage(0, 0);
  for (int t0 = 0; t0 < S_; t0 += 32) {
    const int cur = (t0 >> 5) & 1;
    if (t0 + 32 < S_) { stage(t0 + 32, cur ^ 1); WAIT_ASYNC(10); }
    else              { WAIT_ASYNC(0); }
    __syncthreads();

    // --- scores: S = Q(16x192) * K^T, two 16-col subtiles ---
    v8f s[2] = {};
#pragma unroll
    for (int sub = 0; sub < 2; ++sub)
#pragma unroll
      for (int c = 0; c < 6; ++c) {
        Frag16 bf;
#pragma unroll
        for (int v = 0; v < 8; ++v)
          bf.u[v] = *(const uint32_t*)(&ldsK[cur][(sub * 16 + m_l) * DQK_ +
                                                  c * 32 + (v >> 2) * 16 + half * 8 + (v & 3) * 2]);
        s[sub] = __builtin_amdgcn_wmma_f32_16x16x32_bf16(false, qfr[c].v, false, bf.v,
                                                         (short)0, s[sub], false, false);
      }

    // --- online softmax (rows live in 16-lane halves of the wave) ---
    float p0[8], p1[8], alpha[8];
#pragma unroll
    for (int r = 0; r < 8; ++r) {
      float a = s[0][r] * sc, c2 = s[1][r] * sc;
      float mx = fmaxf(a, c2);
      mx = fmaxf(mx, __shfl_xor(mx, 1, 32));
      mx = fmaxf(mx, __shfl_xor(mx, 2, 32));
      mx = fmaxf(mx, __shfl_xor(mx, 4, 32));
      mx = fmaxf(mx, __shfl_xor(mx, 8, 32));
      float mn = fmaxf(m_run[r], mx);
      alpha[r] = __expf(m_run[r] - mn);
      p0[r] = __expf(a - mn);
      p1[r] = __expf(c2 - mn);
      float rs = p0[r] + p1[r];
      rs += __shfl_xor(rs, 1, 32);
      rs += __shfl_xor(rs, 2, 32);
      rs += __shfl_xor(rs, 4, 32);
      rs += __shfl_xor(rs, 8, 32);
      l_run[r] = l_run[r] * alpha[r] + rs;
      m_run[r] = mn;
    }
#pragma unroll
    for (int t = 0; t < 8; ++t)
#pragma unroll
      for (int r = 0; r < 8; ++r)
        o[t][r] = o[t][r] * alpha[r];

    // --- stage P (16x32 bf16) through per-wave LDS, reload as A fragment ---
#pragma unroll
    for (int r = 0; r < 8; ++r) {
      ldsP[w][r + 8 * half][m_l]      = f2bf(p0[r]);
      ldsP[w][r + 8 * half][16 + m_l] = f2bf(p1[r]);
    }
    Frag16 pa;
#pragma unroll
    for (int v = 0; v < 8; ++v)
      pa.u[v] = *(const uint32_t*)&ldsP[w][m_l][(v >> 2) * 16 + half * 8 + (v & 3) * 2];

    // --- O += P(16x32) * V(32x128), 8 column tiles; V^T tile: token pairs contiguous ---
#pragma unroll
    for (int nc = 0; nc < 8; ++nc) {
      Frag16 vb;
#pragma unroll
      for (int v = 0; v < 8; ++v)
        vb.u[v] = *(const uint32_t*)(&ldsV[cur][(nc * 16 + m_l) * 32 + (v >> 2) * 16 + half * 8 + (v & 3) * 2]);
      o[nc] = __builtin_amdgcn_wmma_f32_16x16x32_bf16(false, pa.v, false, vb.v,
                                                      (short)0, o[nc], false, false);
    }
    __syncthreads();  // all waves done with buffer `cur` before it is refilled
  }

  // --- epilogue: normalize, write bf16 [tok, h*128 + d] ---
  float inv[8];
#pragma unroll
  for (int r = 0; r < 8; ++r) inv[r] = 1.f / l_run[r];
#pragma unroll
  for (int nc = 0; nc < 8; ++nc)
#pragma unroll
    for (int r = 0; r < 8; ++r) {
      size_t row = (size_t)(b * S_ + q0 + r + 8 * half);
      ao[row * (NH_ * DV_) + h * DV_ + nc * 16 + m_l] = f2bf(o[nc][r] * inv[r]);
    }
}

// ---------------- host orchestration ----------------
extern "C" void kernel_launch(void* const* d_in, const int* in_sizes, int n_in,
                              void* d_out, int out_size, void* d_ws, size_t ws_size,
                              hipStream_t stream) {
  const float* x        = (const float*)d_in[0];
  const float* wq_a_w   = (const float*)d_in[1];
  const float* wq_a_b   = (const float*)d_in[2];
  const float* q_norm_w = (const float*)d_in[3];
  const float* wq_b_w   = (const float*)d_in[4];
  const float* wq_b_b   = (const float*)d_in[5];
  const float* wkv_a_w  = (const float*)d_in[6];
  const float* wkv_a_b  = (const float*)d_in[7];
  const float* kv_norm_w= (const float*)d_in[8];
  const float* wkv_b_w  = (const float*)d_in[9];
  const float* wkv_b_b  = (const float*)d_in[10];
  const float* wo_w     = (const float*)d_in[11];
  const float* wo_b     = (const float*)d_in[12];
  float* out = (float*)d_out;

  char* ws = (char*)d_ws;
  size_t off = 0;
  auto alloc = [&](size_t bytes) -> void* {
    off = (off + 255) & ~(size_t)255;
    void* p = ws + off;
    off += bytes;
    return p;
  };

  uint16_t* xb     = (uint16_t*)alloc((size_t)MTOK * DIM_ * 2);
  uint16_t* wqa_h  = (uint16_t*)alloc((size_t)QLR_ * DIM_ * 2);
  uint16_t* wqb_h  = (uint16_t*)alloc((size_t)(NH_ * DQK_) * QLR_ * 2);
  uint16_t* wkva_h = (uint16_t*)alloc((size_t)(KVLR_ + DR_) * DIM_ * 2);
  uint16_t* wkvb_h = (uint16_t*)alloc((size_t)(NH_ * (DN_ + DV_)) * KVLR_ * 2);
  uint16_t* wo_h   = (uint16_t*)alloc((size_t)DIM_ * (NH_ * DV_) * 2);
  uint16_t* qan    = (uint16_t*)alloc((size_t)MTOK * QLR_ * 2);
  float*    kvf    = (float*)   alloc((size_t)MTOK * (KVLR_ + DR_) * 4);
  uint16_t* kvn    = (uint16_t*)alloc((size_t)MTOK * KVLR_ * 2);
  uint16_t* kpe    = (uint16_t*)alloc((size_t)MTOK * DR_ * 2);
  uint16_t* qfb    = (uint16_t*)alloc((size_t)MTOK * NH_ * DQK_ * 2);
  uint16_t* kfb    = (uint16_t*)alloc((size_t)MTOK * NH_ * DQK_ * 2);
  uint16_t* vTb    = (uint16_t*)alloc((size_t)MTOK * NH_ * DV_ * 2);
  uint16_t* aob    = (uint16_t*)alloc((size_t)MTOK * NH_ * DV_ * 2);
  // aliased big region: [qa_f32 24MB][q_f32 48MB]; kb_f32 (64MB) reuses it after both consumed
  char* big = (char*)alloc((size_t)MTOK * QLR_ * 4 + (size_t)MTOK * NH_ * DQK_ * 4);
  float* qaf = (float*)big;
  float* qF  = (float*)(big + (size_t)MTOK * QLR_ * 4);
  float* kbf = (float*)big;

  // 1. bf16 conversions
  k_cvt_bf16<<<1024, 256, 0, stream>>>(x,       xb,     MTOK * DIM_);
  k_cvt_bf16<<<1024, 256, 0, stream>>>(wq_a_w,  wqa_h,  QLR_ * DIM_);
  k_cvt_bf16<<<1024, 256, 0, stream>>>(wq_b_w,  wqb_h,  (NH_ * DQK_) * QLR_);
  k_cvt_bf16<<<1024, 256, 0, stream>>>(wkv_a_w, wkva_h, (KVLR_ + DR_) * DIM_);
  k_cvt_bf16<<<1024, 256, 0, stream>>>(wkv_b_w, wkvb_h, (NH_ * (DN_ + DV_)) * KVLR_);
  k_cvt_bf16<<<1024, 256, 0, stream>>>(wo_w,    wo_h,   DIM_ * (NH_ * DV_));

  // 2. q_a = x @ wq_a^T + b
  k_gemm_bf16<<<dim3(QLR_ / 64, MTOK / 128), 256, 0, stream>>>(
      xb, wqa_h, wq_a_b, qaf, MTOK, QLR_, DIM_);
  // 3. rmsnorm(q_a) -> bf16
  k_rmsnorm_bf16<<<MTOK, 256, 0, stream>>>(qaf, q_norm_w, qan, QLR_, QLR_);
  // 4. q = qan @ wq_b^T + b
  k_gemm_bf16<<<dim3((NH_ * DQK_) / 64, MTOK / 128), 256, 0, stream>>>(
      qan, wqb_h, wq_b_b, qF, MTOK, NH_ * DQK_, QLR_);
  // 5. kv = x @ wkv_a^T + b
  k_gemm_bf16<<<dim3((KVLR_ + DR_) / 64, MTOK / 128), 256, 0, stream>>>(
      xb, wkva_h, wkv_a_b, kvf, MTOK, KVLR_ + DR_, DIM_);
  // 6. rope(k_pe), rmsnorm(kv latent)
  k_rope_kpe<<<(MTOK * DR_ + 255) / 256, 256, 0, stream>>>(kvf, kpe);
  k_rmsnorm_bf16<<<MTOK, 256, 0, stream>>>(kvf, kv_norm_w, kvn, KVLR_, KVLR_ + DR_);
  // 7. assemble qf (before kb gemm overwrites the aliased region)
  k_rope_assemble_q<<<MTOK, 256, 0, stream>>>(qF, qfb);
  // 8. kb = kvn @ wkv_b^T + b   (writes into aliased region)
  k_gemm_bf16<<<dim3((NH_ * (DN_ + DV_)) / 64, MTOK / 128), 256, 0, stream>>>(
      kvn, wkvb_h, wkv_b_b, kbf, MTOK, NH_ * (DN_ + DV_), KVLR_);
  // 9. assemble k (nope + pe) and transposed v
  k_assemble_kv<<<(MTOK * NH_ * DN_ + 255) / 256, 256, 0, stream>>>(kbf, kpe, kfb, vTb);
  // 10. attention
  k_attention<<<B_ * NH_ * (S_ / 64), 128, 0, stream>>>(qfb, kfb, vTb, aob);
  // 11. out = ao @ wo^T + b  (f32 output)
  k_gemm_bf16<<<dim3(DIM_ / 64, MTOK / 128), 256, 0, stream>>>(
      aob, wo_h, wo_b, out, MTOK, DIM_, NH_ * DV_);
}
